// SwinTransformerBlock_1322849927964
// MI455X (gfx1250) — matrix operand order repaired and checked by reading, compiled, not run on hardware
//
#include <hip/hip_runtime.h>
#include <hip/hip_bf16.h>

// ---------------------------------------------------------------------------
// Swin Transformer block (B=64,H=W=56,C=128,WIN=7,SHIFT=3,NH=4,HD=32) for
// gfx1250. Memory-bound (~450MB traffic @23.3TB/s); all GEMMs use
// v_wmma_f32_16x16x32_bf16 (wave32, fp32 accum). Fragment fetches use
// b128 LDS/global loads (layouts are contiguous per lane in 16B runs).
// Kernel 0: convert weights fp32->bf16, expand rel-pos bias table.
// Kernel 1: per-window fused LN1 + shift/partition + QKV + attention +
//           softmax + proj + residual  -> y in d_out.
// Kernel 2: per-64-token fused LN2 + fc1 + GELU + fc2 + residual, in place.
// ---------------------------------------------------------------------------

#define DEVI __device__ __forceinline__

typedef __attribute__((ext_vector_type(16))) __bf16        v16bf;
typedef __attribute__((ext_vector_type(8)))  float         v8f;
typedef __attribute__((ext_vector_type(8)))  unsigned int  v8u;

// problem constants
static constexpr int WIN   = 7;
static constexpr int SHIFT = 3;
static constexpr int NTOK  = 49;      // tokens per window
static constexpr int NHEAD = 4;
static constexpr int CDIM  = 128;
static constexpr int MLPH  = 512;
static constexpr int NWINB = 4096;    // B * nW
static constexpr int TOKTOT= 200704;  // B*H*W

// LDS strides (bf16 elements; all multiples of 8 -> 16B-aligned b128 loads)
static constexpr int AST = 136;   // LN'd tokens     [64][AST]
static constexpr int QST = 40;    // Q/K/V per head  [4][64][QST]
static constexpr int PST = 72;    // per-wave P      [8][16][PST]
static constexpr int OST = 136;   // attention out   [64][OST]
static constexpr int HST = 520;   // MLP hidden      [64][HST]

DEVI unsigned short f2bf(float f) {              // RNE fp32 -> bf16
    unsigned int u = __float_as_uint(f);
    u += 0x7fffu + ((u >> 16) & 1u);
    return (unsigned short)(u >> 16);
}

// A fragment: 16x32 bf16, row-major source [rows][stride].
// Layout: lane = half*16+m holds M=m; VGPR v holds K pair
//   kk = half*16 + 2v  ->  one contiguous 32B run per lane:
//   [k0 + half*16, +16)   => 2x ds_load_b128.
DEVI v8u load_a_frag(const unsigned short* base, int row0, int k0, int stride, int lane) {
    const int m = lane & 15, half = lane >> 4;
    const unsigned short* p = base + (row0 + m) * stride + k0 + (half << 4);
    const uint4 lo = *(const uint4*)p;          // kk = half*16 + 0..7
    const uint4 hi = *(const uint4*)(p + 8);    // kk = half*16 + 8..15
    v8u u;
    u[0] = lo.x; u[1] = lo.y; u[2] = lo.z; u[3] = lo.w;
    u[4] = hi.x; u[5] = hi.y; u[6] = hi.z; u[7] = hi.w;
    return u;
}

// B fragment: 32x16 bf16 (KxN) from an N-major array W[n][k] (k contiguous).
// lane holds N = n0 + (lane&15); VGPR v holds K pair k0 + half*16 + 2v
//   -> one contiguous 32B run per lane => 2x b128 loads.
DEVI v8u load_b_nmajor(const unsigned short* base, int n0, int k0, int stride, int lane) {
    const int n = n0 + (lane & 15), half = lane >> 4;
    const unsigned short* p = base + (size_t)n * stride + k0 + (half << 4);
    const uint4 lo = *(const uint4*)p;
    const uint4 hi = *(const uint4*)(p + 8);
    v8u u;
    u[0] = lo.x; u[1] = lo.y; u[2] = lo.z; u[3] = lo.w;
    u[4] = hi.x; u[5] = hi.y; u[6] = hi.z; u[7] = hi.w;
    return u;
}

// B fragment from a K-major array V[k][n] (n contiguous): K pair = two rows
// (inherently a gather; only used 4x per wave for P@V).
DEVI v8u load_b_kmajor(const unsigned short* base, int k0, int n0, int stride, int lane) {
    v8u u;
    const int n = n0 + (lane & 15), half = lane >> 4;
    const int kb = k0 + (half << 4);
#pragma unroll
    for (int v = 0; v < 8; ++v) {
        unsigned int lo = base[(kb + 2 * v) * stride + n];
        unsigned int hi = base[(kb + 2 * v + 1) * stride + n];
        u[v] = lo | (hi << 16);
    }
    return u;
}

DEVI v8f wmma_bf16(v8u a, v8u b, v8f c) {
    return __builtin_amdgcn_wmma_f32_16x16x32_bf16(
        false, __builtin_bit_cast(v16bf, a),
        false, __builtin_bit_cast(v16bf, b),
        (short)0, c, false, false);
}

// ---------------------------------------------------------------------------
// Kernel 0: weight conversion + rel-pos bias expansion
// ---------------------------------------------------------------------------
__global__ void swin_prep_kernel(
    const float* __restrict__ qkv_w, const float* __restrict__ proj_w,
    const float* __restrict__ fc1_w, const float* __restrict__ fc2_w,
    const float* __restrict__ rpb,
    unsigned short* __restrict__ qkv_wb, unsigned short* __restrict__ proj_wb,
    unsigned short* __restrict__ fc1_wb, unsigned short* __restrict__ fc2_wb,
    float* __restrict__ bias_all)
{
    int i = blockIdx.x * blockDim.x + threadIdx.x;
    if (i < 49152) { qkv_wb[i] = f2bf(qkv_w[i]); return; }
    i -= 49152;
    if (i < 16384) { proj_wb[i] = f2bf(proj_w[i]); return; }
    i -= 16384;
    if (i < 65536) { fc1_wb[i] = f2bf(fc1_w[i]); return; }
    i -= 65536;
    if (i < 65536) { fc2_wb[i] = f2bf(fc2_w[i]); return; }
    i -= 65536;
    if (i < NHEAD * NTOK * NTOK) {
        const int h  = i / (NTOK * NTOK);
        const int ij = i - h * (NTOK * NTOK);
        const int qi = ij / NTOK, kj = ij - qi * NTOK;
        const int hi = qi / WIN, wi = qi - hi * WIN;
        const int hj = kj / WIN, wj = kj - hj * WIN;
        const int idx = (hi - hj + WIN - 1) * (2 * WIN - 1) + (wi - wj + WIN - 1);
        bias_all[i] = rpb[idx * NHEAD + h];
    }
}

// ---------------------------------------------------------------------------
// Kernel 1: fused window attention (one workgroup = one window, 8 waves)
// ---------------------------------------------------------------------------
__global__ __launch_bounds__(256) void swin_attn_kernel(
    const float* __restrict__ x,
    const float* __restrict__ ln1_g, const float* __restrict__ ln1_b,
    const unsigned short* __restrict__ qkv_wb, const float* __restrict__ qkv_b,
    const unsigned short* __restrict__ proj_wb, const float* __restrict__ proj_b,
    const float* __restrict__ bias_all,
    float* __restrict__ y)
{
    extern __shared__ unsigned short sm1[];
    unsigned short* sA = sm1;                       // 64*AST
    unsigned short* sQ = sA + 64 * AST;             // 4*64*QST
    unsigned short* sK = sQ + NHEAD * 64 * QST;
    unsigned short* sV = sK + NHEAD * 64 * QST;
    unsigned short* sP = sV + NHEAD * 64 * QST;     // 8*16*PST
    unsigned short* sO = sP + 8 * 16 * PST;         // 64*OST

    const int tid  = threadIdx.x;
    const int wid  = tid >> 5, lane = tid & 31;
    const int win  = blockIdx.x;
    const int bimg = win >> 6;
    const int wh   = (win >> 3) & 7, ww = win & 7;

    // pull weights toward L2 (global_prefetch_b8)
    if (tid < 128) {
        __builtin_prefetch(qkv_wb + tid * 384, 0, 0);
        __builtin_prefetch(proj_wb + tid * 128, 0, 0);
    }

    // ---- LN1 + cyclic shift + window partition (4 lanes per token) ----
    {
        const int tok = tid >> 2, part = tid & 3;
        float vals[32];
        float s = 0.f, ss = 0.f;
        if (tok < NTOK) {
            const int hi = tok / WIN, wi = tok - hi * WIN;
            int h = wh * WIN + hi + SHIFT; if (h >= 56) h -= 56;
            int w = ww * WIN + wi + SHIFT; if (w >= 56) w -= 56;
            const float4* p4 = (const float4*)(x + ((size_t)(bimg * 3136 + h * 56 + w)) * CDIM + part * 32);
#pragma unroll
            for (int j = 0; j < 8; ++j) {
                float4 q4 = p4[j];
                vals[4 * j + 0] = q4.x; vals[4 * j + 1] = q4.y;
                vals[4 * j + 2] = q4.z; vals[4 * j + 3] = q4.w;
                s  += q4.x + q4.y + q4.z + q4.w;
                ss += q4.x * q4.x + q4.y * q4.y + q4.z * q4.z + q4.w * q4.w;
            }
        }
        s  += __shfl_xor(s, 1);  s  += __shfl_xor(s, 2);
        ss += __shfl_xor(ss, 1); ss += __shfl_xor(ss, 2);
        const float mean = s * (1.f / 128.f);
        const float rstd = rsqrtf(ss * (1.f / 128.f) - mean * mean + 1e-5f);
        unsigned short* row = sA + tok * AST + part * 32;
        if (tok < NTOK) {
#pragma unroll
            for (int i = 0; i < 32; ++i) {
                const int c = part * 32 + i;
                row[i] = f2bf((vals[i] - mean) * rstd * ln1_g[c] + ln1_b[c]);
            }
        } else {
#pragma unroll
            for (int i = 0; i < 32; ++i) row[i] = 0;
        }
    }
    __syncthreads();

    // ---- QKV GEMM: [64x128] @ [128x384] -> head-split Q(scaled)/K/V in LDS ----
    {
        const int col = lane & 15, half = lane >> 4;
#pragma unroll 1
        for (int it = 0; it < 12; ++it) {
            const int tile = wid * 12 + it;          // 96 tiles = 4 Mt x 24 Nt
            const int mt = tile / 24, nt = tile - mt * 24;
            v8f acc = {};
#pragma unroll
            for (int ks = 0; ks < 4; ++ks) {
                v8u a  = load_a_frag(sA, mt * 16, ks * 32, AST, lane);
                v8u bb = load_b_nmajor(qkv_wb, nt * 16, ks * 32, CDIM, lane);
                acc = wmma_bf16(a, bb, acc);
            }
            const int f     = nt * 16 + col;
            const float qb  = qkv_b[f];
            const int which = f >> 7;                // 0=q 1=k 2=v
            const int ff    = f & 127;
            const int head  = ff >> 5, hd = ff & 31;
            unsigned short* dst =
                (which == 0 ? sQ : (which == 1 ? sK : sV)) + head * 64 * QST + hd;
            const float scale = (which == 0) ? 0.17677669529663687f : 1.0f; // HD^-0.5
#pragma unroll
            for (int r = 0; r < 8; ++r)
                dst[(mt * 16 + half * 8 + r) * QST] = f2bf((acc[r] + qb) * scale);
        }
    }
    __syncthreads();

    // ---- attention: wave -> (head = wid&3, M-half = wid>>2) ----
    {
        const int head = wid & 3;
        const int mg   = wid >> 2;
        const int col  = lane & 15, half = lane >> 4;
        const unsigned short* Qh = sQ + head * 64 * QST;
        const unsigned short* Kh = sK + head * 64 * QST;
        const unsigned short* Vh = sV + head * 64 * QST;
        unsigned short* Pw = sP + wid * 16 * PST;
        const float* biasH = bias_all + head * (NTOK * NTOK);

#pragma unroll 1
        for (int mi = 0; mi < 2; ++mi) {
            const int mt = mg * 2 + mi;
            v8f sv[4];
            {   // S = Q @ K^T  (K-dim = HD = 32 -> one WMMA per tile)
                v8u qa = load_a_frag(Qh, mt * 16, 0, QST, lane);
#pragma unroll
                for (int nt = 0; nt < 4; ++nt) {
                    v8u kb = load_b_nmajor(Kh, nt * 16, 0, QST, lane);
                    v8f z = {};
                    sv[nt] = wmma_bf16(qa, kb, z);
                }
            }
            // + rel-pos bias + SW-MSA mask; mask padded keys to -inf
#pragma unroll
            for (int nt = 0; nt < 4; ++nt) {
#pragma unroll
                for (int r = 0; r < 8; ++r) {
                    const int rowi = mt * 16 + half * 8 + r;
                    const int cc   = nt * 16 + col;
                    float v = sv[nt][r];
                    if (rowi < NTOK && cc < NTOK) {
                        v += biasH[rowi * NTOK + cc];
                        const int hi = rowi / WIN, wi = rowi - hi * WIN;
                        const int hj = cc / WIN,  wj = cc - hj * WIN;
                        const int li = ((wh == 7) ? (hi < WIN - SHIFT ? 1 : 2) : 0) * 3
                                     + ((ww == 7) ? (wi < WIN - SHIFT ? 1 : 2) : 0);
                        const int lj = ((wh == 7) ? (hj < WIN - SHIFT ? 1 : 2) : 0) * 3
                                     + ((ww == 7) ? (wj < WIN - SHIFT ? 1 : 2) : 0);
                        if (li != lj) v -= 100.f;
                    } else {
                        v = -1e30f;
                    }
                    sv[nt][r] = v;
                }
            }
            // row softmax: rows live across a 16-lane half-wave
            float inv[8];
#pragma unroll
            for (int r = 0; r < 8; ++r) {
                float m = fmaxf(fmaxf(sv[0][r], sv[1][r]), fmaxf(sv[2][r], sv[3][r]));
                m = fmaxf(m, __shfl_xor(m, 1));
                m = fmaxf(m, __shfl_xor(m, 2));
                m = fmaxf(m, __shfl_xor(m, 4));
                m = fmaxf(m, __shfl_xor(m, 8));
                const float e0 = __expf(sv[0][r] - m), e1 = __expf(sv[1][r] - m);
                const float e2 = __expf(sv[2][r] - m), e3 = __expf(sv[3][r] - m);
                sv[0][r] = e0; sv[1][r] = e1; sv[2][r] = e2; sv[3][r] = e3;
                float su = e0 + e1 + e2 + e3;
                su += __shfl_xor(su, 1);
                su += __shfl_xor(su, 2);
                su += __shfl_xor(su, 4);
                su += __shfl_xor(su, 8);
                inv[r] = 1.0f / su;
            }
            // write P (bf16) to this wave's LDS tile
#pragma unroll
            for (int nt = 0; nt < 4; ++nt)
#pragma unroll
                for (int r = 0; r < 8; ++r)
                    Pw[(half * 8 + r) * PST + nt * 16 + col] = f2bf(sv[nt][r] * inv[r]);
            // same-wave LDS RAW: P stores -> A-fragment loads
            asm volatile("s_wait_dscnt 0" ::: "memory");
            // O = P @ V  (K = 64 keys -> 2 steps; N = HD -> 2 tiles)
#pragma unroll
            for (int nt2 = 0; nt2 < 2; ++nt2) {
                v8f oacc = {};
#pragma unroll
                for (int ks = 0; ks < 2; ++ks) {
                    v8u pa = load_a_frag(Pw, 0, ks * 32, PST, lane);
                    v8u vb = load_b_kmajor(Vh, ks * 32, nt2 * 16, QST, lane);
                    oacc = wmma_bf16(pa, vb, oacc);
                }
#pragma unroll
                for (int r = 0; r < 8; ++r)
                    sO[(mt * 16 + half * 8 + r) * OST + head * 32 + nt2 * 16 + col] =
                        f2bf(oacc[r]);
            }
        }
    }
    __syncthreads();

    // ---- proj GEMM + window reverse + reverse shift + residual -> y ----
    {
        const int col = lane & 15, half = lane >> 4;
#pragma unroll 1
        for (int it = 0; it < 4; ++it) {
            const int tile = wid * 4 + it;           // 32 tiles = 4 Mt x 8 Nt
            const int mt = tile >> 3, nt = tile & 7;
            v8f acc = {};
#pragma unroll
            for (int ks = 0; ks < 4; ++ks) {
                v8u a  = load_a_frag(sO, mt * 16, ks * 32, OST, lane);
                v8u bb = load_b_nmajor(proj_wb, nt * 16, ks * 32, CDIM, lane);
                acc = wmma_bf16(a, bb, acc);
            }
            const int c   = nt * 16 + col;
            const float pb = proj_b[c];
#pragma unroll
            for (int r = 0; r < 8; ++r) {
                const int rowi = mt * 16 + half * 8 + r;
                if (rowi < NTOK) {
                    const int hi = rowi / WIN, wi = rowi - hi * WIN;
                    int h = wh * WIN + hi + SHIFT; if (h >= 56) h -= 56;
                    int w = ww * WIN + wi + SHIFT; if (w >= 56) w -= 56;
                    const size_t gi = ((size_t)(bimg * 3136 + h * 56 + w)) * CDIM + c;
                    y[gi] = x[gi] + acc[r] + pb;
                }
            }
        }
    }
}

// ---------------------------------------------------------------------------
// Kernel 2: fused MLP, in place on d_out (one workgroup = 64 tokens)
// ---------------------------------------------------------------------------
__global__ __launch_bounds__(256) void swin_mlp_kernel(
    const float* __restrict__ ln2_g, const float* __restrict__ ln2_b,
    const unsigned short* __restrict__ fc1_wb, const float* __restrict__ fc1_b,
    const unsigned short* __restrict__ fc2_wb, const float* __restrict__ fc2_b,
    float* __restrict__ y)
{
    extern __shared__ unsigned short sm2[];
    unsigned short* sA = sm2;             // 64*AST
    unsigned short* sH = sA + 64 * AST;   // 64*HST

    const int tid  = threadIdx.x;
    const int wid  = tid >> 5, lane = tid & 31;
    const int col  = lane & 15, half = lane >> 4;
    const int tok0 = blockIdx.x * 64;

    __builtin_prefetch(fc1_wb + tid * 256, 0, 0);
    __builtin_prefetch(fc2_wb + tid * 256, 0, 0);

    // ---- LN2 ----
    {
        const int tok = tid >> 2, part = tid & 3;
        const float4* p4 = (const float4*)(y + ((size_t)(tok0 + tok)) * CDIM + part * 32);
        float vals[32];
        float s = 0.f, ss = 0.f;
#pragma unroll
        for (int j = 0; j < 8; ++j) {
            float4 q4 = p4[j];
            vals[4 * j + 0] = q4.x; vals[4 * j + 1] = q4.y;
            vals[4 * j + 2] = q4.z; vals[4 * j + 3] = q4.w;
            s  += q4.x + q4.y + q4.z + q4.w;
            ss += q4.x * q4.x + q4.y * q4.y + q4.z * q4.z + q4.w * q4.w;
        }
        s  += __shfl_xor(s, 1);  s  += __shfl_xor(s, 2);
        ss += __shfl_xor(ss, 1); ss += __shfl_xor(ss, 2);
        const float mean = s * (1.f / 128.f);
        const float rstd = rsqrtf(ss * (1.f / 128.f) - mean * mean + 1e-5f);
        unsigned short* row = sA + tok * AST + part * 32;
#pragma unroll
        for (int i = 0; i < 32; ++i) {
            const int c = part * 32 + i;
            row[i] = f2bf((vals[i] - mean) * rstd * ln2_g[c] + ln2_b[c]);
        }
    }
    __syncthreads();

    // ---- fc1 + exact GELU: [64x128] @ [128x512] ----
#pragma unroll 1
    for (int it = 0; it < 16; ++it) {
        const int tile = wid * 16 + it;              // 128 tiles = 4 Mt x 32 Nt
        const int mt = tile >> 5, nt = tile & 31;
        v8f acc = {};
#pragma unroll
        for (int ks = 0; ks < 4; ++ks) {
            v8u a  = load_a_frag(sA, mt * 16, ks * 32, AST, lane);
            v8u bb = load_b_nmajor(fc1_wb, nt * 16, ks * 32, CDIM, lane);
            acc = wmma_bf16(a, bb, acc);
        }
        const int f   = nt * 16 + col;
        const float fb = fc1_b[f];
#pragma unroll
        for (int r = 0; r < 8; ++r) {
            const float v = acc[r] + fb;
            const float g = 0.5f * v * (1.f + erff(v * 0.70710678118654752f));
            sH[(mt * 16 + half * 8 + r) * HST + f] = f2bf(g);
        }
    }
    __syncthreads();

    // ---- fc2 + residual: [64x512] @ [512x128] ----
#pragma unroll 1
    for (int it = 0; it < 4; ++it) {
        const int tile = wid * 4 + it;               // 32 tiles = 4 Mt x 8 Nt
        const int mt = tile >> 3, nt = tile & 7;
        v8f acc = {};
#pragma unroll
        for (int ks = 0; ks < 16; ++ks) {
            v8u a  = load_a_frag(sH, mt * 16, ks * 32, HST, lane);
            v8u bb = load_b_nmajor(fc2_wb, nt * 16, ks * 32, MLPH, lane);
            acc = wmma_bf16(a, bb, acc);
        }
        const int c   = nt * 16 + col;
        const float cb = fc2_b[c];
#pragma unroll
        for (int r = 0; r < 8; ++r) {
            const int rowi = mt * 16 + half * 8 + r;
            const size_t gi = ((size_t)(tok0 + rowi)) * CDIM + c;
            y[gi] = y[gi] + acc[r] + cb;             // in-place residual
        }
    }
}

// ---------------------------------------------------------------------------
extern "C" void kernel_launch(void* const* d_in, const int* in_sizes, int n_in,
                              void* d_out, int out_size, void* d_ws, size_t ws_size,
                              hipStream_t stream)
{
    const float* x      = (const float*)d_in[0];
    const float* ln1_g  = (const float*)d_in[1];
    const float* ln1_b  = (const float*)d_in[2];
    const float* qkv_w  = (const float*)d_in[3];
    const float* qkv_b  = (const float*)d_in[4];
    const float* rpb    = (const float*)d_in[5];
    const float* proj_w = (const float*)d_in[6];
    const float* proj_b = (const float*)d_in[7];
    const float* ln2_g  = (const float*)d_in[8];
    const float* ln2_b  = (const float*)d_in[9];
    const float* fc1_w  = (const float*)d_in[10];
    const float* fc1_b  = (const float*)d_in[11];
    const float* fc2_w  = (const float*)d_in[12];
    const float* fc2_b  = (const float*)d_in[13];
    float* y = (float*)d_out;

    // workspace: bf16 weights + expanded bias (~432 KB)
    unsigned short* wsb     = (unsigned short*)d_ws;
    unsigned short* qkv_wb  = wsb;               // 49152 u16
    unsigned short* proj_wb = wsb + 49152;       // 16384 u16
    unsigned short* fc1_wb  = wsb + 65536;       // 65536 u16
    unsigned short* fc2_wb  = wsb + 131072;      // 65536 u16
    float*          bias_all= (float*)(wsb + 196608); // 9604 f32

    const int prep_n = 49152 + 16384 + 65536 + 65536 + NHEAD * NTOK * NTOK;
    swin_prep_kernel<<<(prep_n + 255) / 256, 256, 0, stream>>>(
        qkv_w, proj_w, fc1_w, fc2_w, rpb,
        qkv_wb, proj_wb, fc1_wb, fc2_wb, bias_all);

    const size_t smem1 = (size_t)(64 * AST + 3 * NHEAD * 64 * QST + 8 * 16 * PST + 64 * OST)
                         * sizeof(unsigned short);   // ~115 KB (<< 320 KB/WGP)
    swin_attn_kernel<<<NWINB, 256, smem1, stream>>>(
        x, ln1_g, ln1_b, qkv_wb, qkv_b, proj_wb, proj_b, bias_all, y);

    const size_t smem2 = (size_t)(64 * AST + 64 * HST) * sizeof(unsigned short); // ~84 KB
    swin_mlp_kernel<<<TOKTOT / 64, 256, smem2, stream>>>(
        ln2_g, ln2_b, fc1_wb, fc1_b, fc2_wb, fc2_b, y);
}
